// MultiHeadCrossAttention_35338990911901
// MI455X (gfx1250) — matrix operand order repaired and checked by reading, not compiled, not run on hardware
//
#include <hip/hip_runtime.h>
#include <stdint.h>

// ---- CDNA5 WMMA types ----
typedef __attribute__((ext_vector_type(16))) __bf16 v16bf;
typedef __attribute__((ext_vector_type(8)))  float  v8f;

#define DEV static __device__ __forceinline__

union BFV { v16bf v; uint4 q[2]; unsigned short us[16]; };

DEV unsigned short f2bfu(float f) {
  unsigned u = __float_as_uint(f);
  unsigned r = u + 0x7FFFu + ((u >> 16) & 1u);   // round-to-nearest-even
  return (unsigned short)(r >> 16);
}

DEV v8f wmma_bf16(v16bf a, v16bf b, v8f c) {
  return __builtin_amdgcn_wmma_f32_16x16x32_bf16(
      /*neg_a=*/false, a, /*neg_b=*/false, b,
      /*c_mod=*/(short)0, c, /*reuse_a=*/false, /*reuse_b=*/false);
}

// A operand (16x32 bf16), source row-major MxK bf16, K contiguous.
// ISA layout: lane holds row M=lane&15; g=lane>>4 selects K sub-halves
// (elems 0..7 = K g*8..g*8+7, elems 8..15 = K 16+g*8..16+g*8+7).
DEV v16bf load_a_bf16(const unsigned short* src, int stride, int lane) {
  int g = lane >> 4, r = lane & 15;
  const unsigned short* p = src + r * stride;
  BFV a;
  a.q[0] = *(const uint4*)(p + g * 8);
  a.q[1] = *(const uint4*)(p + 16 + g * 8);
  return a.v;
}

// B operand (32x16), source is B^T row-major NxK bf16 (K contiguous).
// ISA layout: lane n holds column n&15; g=lane>>4: elems 0..7 = K g*8..g*8+7, elems 8..15 = K 16+g*8..16+g*8+7.
DEV v16bf load_bt_bf16(const unsigned short* srcT, int stride, int lane) {
  int g = lane >> 4, n = lane & 15;
  const unsigned short* p = srcT + n * stride;
  BFV b;
  b.q[0] = *(const uint4*)(p + g * 8);
  b.q[1] = *(const uint4*)(p + 16 + g * 8);
  return b.v;
}

// Problem constants
static constexpr int Bsz = 2, H = 16, S = 2048, D = 1024, DH = 64;
static constexpr int R = Bsz * S;              // 4096 token rows
static constexpr size_t M4 = (size_t)R * D;    // 4M elements
static constexpr size_t M1 = (size_t)D * D;    // 1M elements

// ---------------- Kernel 0: one-shot fp32 -> bf16 conversion ----------------
// Each element converted exactly once (vs 16x-256x if done inside GEMM loops).
__global__ __launch_bounds__(256) void cvt_bf16_kernel(
    const float* __restrict__ Q, const float* __restrict__ K,
    const float* __restrict__ V, const float* __restrict__ Wq,
    const float* __restrict__ Wk, const float* __restrict__ Wv,
    const float* __restrict__ Wo, unsigned short* __restrict__ ws) {
  int z = blockIdx.y;
  const float* src;
  unsigned short* dst;
  size_t n8;  // number of 8-element groups
  switch (z) {
    case 0: src = Q;  dst = ws;                 n8 = M4 / 8; break;
    case 1: src = K;  dst = ws + M4;            n8 = M4 / 8; break;
    case 2: src = V;  dst = ws + 2 * M4;        n8 = M4 / 8; break;
    case 3: src = Wq; dst = ws + 3 * M4;        n8 = M1 / 8; break;
    case 4: src = Wk; dst = ws + 3 * M4 + M1;   n8 = M1 / 8; break;
    case 5: src = Wv; dst = ws + 3 * M4 + 2*M1; n8 = M1 / 8; break;
    default:src = Wo; dst = ws + 3 * M4 + 3*M1; n8 = M1 / 8; break;
  }
  size_t i = (size_t)blockIdx.x * 256 + threadIdx.x;
  if (i < n8) {
    float4 f0 = ((const float4*)src)[2 * i];
    float4 f1 = ((const float4*)src)[2 * i + 1];
    union { unsigned short us[8]; uint4 q; } o;
    o.us[0] = f2bfu(f0.x); o.us[1] = f2bfu(f0.y);
    o.us[2] = f2bfu(f0.z); o.us[3] = f2bfu(f0.w);
    o.us[4] = f2bfu(f1.x); o.us[5] = f2bfu(f1.y);
    o.us[6] = f2bfu(f1.z); o.us[7] = f2bfu(f1.w);
    ((uint4*)dst)[i] = o.q;
  }
}

// ---------------- Kernel 1: QKV projections (y = X @ W^T), bf16 in/out ------
// 32x64 output tile per wave: 2 A operands x 4 B operands = 8 WMMAs/k-step.
__global__ __launch_bounds__(128) void proj_qkv_kernel(
    const unsigned short* __restrict__ Xq, const unsigned short* __restrict__ Xk,
    const unsigned short* __restrict__ Xv, const unsigned short* __restrict__ Wqb,
    const unsigned short* __restrict__ Wkb, const unsigned short* __restrict__ Wvb,
    unsigned short* qh, unsigned short* kh, unsigned short* vT) {
  int z = blockIdx.y;
  const unsigned short* X = (z == 0) ? Xq : (z == 1) ? Xk : Xv;
  const unsigned short* W = (z == 0) ? Wqb : (z == 1) ? Wkb : Wvb;
  int lane = threadIdx.x & 31, wave = threadIdx.x >> 5;
  int tid = blockIdx.x * 4 + wave;        // 0..2047 tiles
  int mt = tid >> 4, nt = tid & 15;       // 128 row-tiles x 16 col-tiles
  int m0 = mt * 32, n0 = nt * 64;

  v8f acc[2][4] = {};
  for (int k0 = 0; k0 < D; k0 += 32) {
    v16bf A0 = load_a_bf16(X + m0 * D + k0, D, lane);
    v16bf A1 = load_a_bf16(X + (m0 + 16) * D + k0, D, lane);
#pragma unroll
    for (int t = 0; t < 4; ++t) {
      v16bf Bop = load_bt_bf16(W + (n0 + t * 16) * D + k0, D, lane);
      acc[0][t] = wmma_bf16(A0, Bop, acc[0][t]);
      acc[1][t] = wmma_bf16(A1, Bop, acc[1][t]);
    }
  }
  // Scatter C-layout tiles: lane carries (M = v + 8*(lane>>4), N = lane&15)
  int g = lane >> 4, n = lane & 15;
#pragma unroll
  for (int half = 0; half < 2; ++half) {
#pragma unroll
    for (int t = 0; t < 4; ++t) {
      int f = n0 + t * 16 + n;          // output feature
      int h = f >> 6, dd = f & 63;      // head, dim-in-head
#pragma unroll
      for (int v = 0; v < 8; ++v) {
        int m = m0 + half * 16 + v + 8 * g;
        int b = m >> 11, s = m & (S - 1);
        unsigned short val = f2bfu(acc[half][t][v]);
        if (z < 2) {
          ((z == 0) ? qh : kh)[((b * H + h) * S + s) * DH + dd] = val;  // [B,H,S,64]
        } else {
          vT[((b * H + h) * DH + dd) * S + s] = val;                    // [B,H,64,S]
        }
      }
    }
  }
}

// ---------------- Kernel 2: attention (streaming softmax, no max-rescale; ---
// legal since reference clamps scores/8 to +-50 => exp <= 5.2e21, f32-safe).
__global__ __launch_bounds__(128) void attn_kernel(
    const unsigned short* __restrict__ qh, const unsigned short* __restrict__ kh,
    const unsigned short* __restrict__ vT, const unsigned char* __restrict__ mask,
    unsigned short* __restrict__ ao) {
  __shared__ __align__(16) unsigned short Plds[4][16 * 32];  // per-wave P tile
  int lane = threadIdx.x & 31, wave = threadIdx.x >> 5;
  int wid = blockIdx.x * 4 + wave;            // 0..4095
  int qt = wid & 127, h = (wid >> 7) & 15, b = wid >> 11;
  int q0 = qt * 16;
  int bh = b * H + h;
  const unsigned short* qbase = qh + (bh * S + q0) * DH;
  const unsigned short* kbase = kh + bh * S * DH;
  const unsigned short* vbase = vT + bh * DH * S;

  // 16x64 query tile kept in registers (two K=32 A operands)
  v16bf qa0 = load_a_bf16(qbase + 0, DH, lane);
  v16bf qa1 = load_a_bf16(qbase + 32, DH, lane);

  v8f oacc[4] = {};
  float rs[8] = {0.f, 0.f, 0.f, 0.f, 0.f, 0.f, 0.f, 0.f};
  int g = lane >> 4, n = lane & 15, r = lane & 15;
  unsigned short* pl = Plds[wave];

  // exp(s/8) == exp2(s * log2(e)/8); clamp(s/8, +-50) == clamp(arg, +-50*log2e)
  const float C  = 0.18033688011112042f;   // log2(e)/8
  const float CL = 72.13475204444817f;     // 50*log2(e)

  for (int kblk = 0; kblk < S; kblk += 32) {
    if (kblk + 32 < S)
      __builtin_prefetch(kbase + (kblk + 32) * DH, 0, 1);   // global_prefetch_b8

    // scores: two 16x16 subtiles, K=64 (2 WMMAs each)
    v8f s0 = {}, s1 = {};
    v16bf b00 = load_bt_bf16(kbase + kblk * DH + 0, DH, lane);
    v16bf b01 = load_bt_bf16(kbase + kblk * DH + 32, DH, lane);
    s0 = wmma_bf16(qa0, b00, s0);
    s0 = wmma_bf16(qa1, b01, s0);
    v16bf b10 = load_bt_bf16(kbase + (kblk + 16) * DH + 0, DH, lane);
    v16bf b11 = load_bt_bf16(kbase + (kblk + 16) * DH + 32, DH, lane);
    s1 = wmma_bf16(qa0, b10, s1);
    s1 = wmma_bf16(qa1, b11, s1);

    // clamp/exp/row-sum; write bf16 P (16 rows x 32 keys) to this wave's LDS
#pragma unroll
    for (int v = 0; v < 8; ++v) {
      float e0 = exp2f(fminf(fmaxf(s0[v] * C, -CL), CL));
      float e1 = exp2f(fminf(fmaxf(s1[v] * C, -CL), CL));
      rs[v] += e0 + e1;
      int row = v + 8 * g;
      pl[row * 32 + n]      = f2bfu(e0);
      pl[row * 32 + 16 + n] = f2bfu(e1);
    }

    // read P back in A layout (compiler inserts the ds-cnt waits)
    BFV pa;
    pa.q[0] = *(const uint4*)&pl[r * 32 + g * 8];
    pa.q[1] = *(const uint4*)&pl[r * 32 + 16 + g * 8];

    // out += P(16x32) @ V(32x64) via 4 WMMAs; vT rows are K-contiguous
#pragma unroll
    for (int t = 0; t < 4; ++t) {
      v16bf bv = load_bt_bf16(vbase + (t * 16) * S + kblk, S, lane);
      oacc[t] = wmma_bf16(pa.v, bv, oacc[t]);
    }
  }

  // row-sum reduction across 16-lane halves (rows 0-7 live in lanes 0-15,
  // rows 8-15 in lanes 16-31 for the C layout)
#pragma unroll
  for (int v = 0; v < 8; ++v) {
    float t = rs[v];
    t += __shfl_xor(t, 1);
    t += __shfl_xor(t, 2);
    t += __shfl_xor(t, 4);
    t += __shfl_xor(t, 8);
    rs[v] = t;
  }

  const unsigned char* mrow = mask + b * S + q0;   // [B,Sq] broadcast over keys
#pragma unroll
  for (int v = 0; v < 8; ++v) {
    int row = v + 8 * g;
    float denom = (rs[v] <= 0.f) ? 1.f : rs[v];
    float scale = mrow[row] ? (1.f / denom) : 0.f;  // masked row -> zero output
#pragma unroll
    for (int t = 0; t < 4; ++t) {
      int qi = q0 + row;
      ao[(b * S + qi) * D + h * DH + t * 16 + n] = f2bfu(oacc[t][v] * scale);
    }
  }
}

// ---------------- Kernel 3: output projection, fp32 result ------------------
__global__ __launch_bounds__(128) void proj_o_kernel(
    const unsigned short* __restrict__ ao, const unsigned short* __restrict__ Wob,
    float* __restrict__ out) {
  int lane = threadIdx.x & 31, wave = threadIdx.x >> 5;
  int tid = blockIdx.x * 4 + wave;        // 0..2047
  int mt = tid >> 4, nt = tid & 15;
  int m0 = mt * 32, n0 = nt * 64;

  v8f acc[2][4] = {};
  for (int k0 = 0; k0 < D; k0 += 32) {
    v16bf A0 = load_a_bf16(ao + m0 * D + k0, D, lane);
    v16bf A1 = load_a_bf16(ao + (m0 + 16) * D + k0, D, lane);
#pragma unroll
    for (int t = 0; t < 4; ++t) {
      v16bf Bop = load_bt_bf16(Wob + (n0 + t * 16) * D + k0, D, lane);
      acc[0][t] = wmma_bf16(A0, Bop, acc[0][t]);
      acc[1][t] = wmma_bf16(A1, Bop, acc[1][t]);
    }
  }
  int g = lane >> 4, n = lane & 15;
#pragma unroll
  for (int half = 0; half < 2; ++half) {
#pragma unroll
    for (int t = 0; t < 4; ++t) {
#pragma unroll
      for (int v = 0; v < 8; ++v) {
        int m = m0 + half * 16 + v + 8 * g;
        out[m * D + n0 + t * 16 + n] = acc[half][t][v];
      }
    }
  }
}

extern "C" void kernel_launch(void* const* d_in, const int* in_sizes, int n_in,
                              void* d_out, int out_size, void* d_ws, size_t ws_size,
                              hipStream_t stream) {
  const float* Q  = (const float*)d_in[0];
  const float* K  = (const float*)d_in[1];
  const float* V  = (const float*)d_in[2];
  const unsigned char* mask = (const unsigned char*)d_in[3];
  const float* Wq = (const float*)d_in[4];
  const float* Wk = (const float*)d_in[5];
  const float* Wv = (const float*)d_in[6];
  const float* Wo = (const float*)d_in[7];

  // workspace layout (ushort elements), 32M elements = 64 MB total:
  // [Qb Kb Vb | Wqb Wkb Wvb Wob | qh kh vT ao]
  unsigned short* ws  = (unsigned short*)d_ws;
  unsigned short* Qb  = ws;
  unsigned short* Kb  = Qb + M4;
  unsigned short* Vb  = Kb + M4;
  unsigned short* Wqb = Vb + M4;
  unsigned short* Wkb = Wqb + M1;
  unsigned short* Wvb = Wkb + M1;
  unsigned short* Wob = Wvb + M1;
  unsigned short* qh  = Wob + M1;
  unsigned short* kh  = qh + M4;
  unsigned short* vT  = kh + M4;
  unsigned short* ao  = vT + M4;

  // 0) one-shot bf16 conversion (bandwidth bound, each element once)
  dim3 cgrid((unsigned)((M4 / 8 + 255) / 256), 7);
  cvt_bf16_kernel<<<cgrid, 256, 0, stream>>>(Q, K, V, Wq, Wk, Wv, Wo, ws);

  // 1) QKV projections: 2048 wave-tiles per GEMM, 4 waves/block
  dim3 pgrid(512, 3);
  proj_qkv_kernel<<<pgrid, 128, 0, stream>>>(Qb, Kb, Vb, Wqb, Wkb, Wvb, qh, kh, vT);

  // 2) attention: 4096 wave-tiles
  attn_kernel<<<1024, 128, 0, stream>>>(qh, kh, vT, mask, ao);

  // 3) output projection
  proj_o_kernel<<<512, 128, 0, stream>>>(ao, Wob, (float*)d_out);
}
